// Decoder_85942295593564
// MI455X (gfx1250) — matrix operand order repaired and checked by reading, not compiled, not run on hardware
//
#include <hip/hip_runtime.h>

// ---------------------------------------------------------------------------
// LAS attention decoder for MI455X (gfx1250), wave32, WMMA bf16.
// T=400, N=256, KS=VS=128, H=512, VOCAB=1000, TL=250.
// One persistent workgroup per 16-row batch tile (16 WGs), 512 threads
// (16 waves). All GEMMs via v_wmma_f32_16x16x32_bf16; recurrence state in LDS.
// GEMM1 runs as two 4-tile passes through an LDS preactivation buffer to keep
// register pressure low (avoids B-fragment spills to scratch).
// ---------------------------------------------------------------------------

typedef __bf16 bf16_t;
typedef __attribute__((ext_vector_type(16))) __bf16 bf16x16;
typedef __attribute__((ext_vector_type(8)))  __bf16 bf16x8;
typedef __attribute__((ext_vector_type(8)))  float  f32x8;
typedef __attribute__((ext_vector_type(4)))  float  f32x4;

#define T_ENC   400
#define NB      256
#define KSZ     128
#define VSZ     128
#define HID     512
#define VOCAB   1000
#define TLEN    250
#define VPAD    1008   // VOCAB padded to 63 tiles of 16

// K-widths of the concatenated GEMM inputs
#define K1      1152   // [emb(512) | ctx(128) | h1(512)]
#define K2      640    // [h1(512) | h2(128)]
#define K3      256    // [h2(128) | ctx(128)]

// ---------------------------------------------------------------------------
// WMMA helpers (contiguous k order K = kk + 16*hi + e in both A and B)
// ---------------------------------------------------------------------------
__device__ __forceinline__ f32x8 wmma_bf(bf16x16 a, bf16x16 b, f32x8 c) {
  return __builtin_amdgcn_wmma_f32_16x16x32_bf16(
      /*neg_a=*/false, a, /*neg_b=*/false, b,
      /*c_mod=*/(short)0, c, /*reuse_a=*/false, /*reuse_b=*/false);
}

__device__ __forceinline__ f32x8 zero8() {
  f32x8 z;
#pragma unroll
  for (int r = 0; r < 8; ++r) z[r] = 0.0f;
  return z;
}

// A fragment: 16x32 bf16 (MxK), row-major [16][ldk] in LDS.
// lane = (l, hi): M = l; elem e: K = kk + 16*hi + e
// -> one contiguous 32B load at kk+16*hi.
__device__ __forceinline__ bf16x16 load_a(const bf16_t* x, int ldk, int kk,
                                          int l, int hi) {
  const bf16_t* p = x + l * ldk + kk + 16 * hi;
  return *(const bf16x16*)(p);
}

// B fragment: 32x16 bf16 (KxN) from row-major weights W[ncols][ldk]
// (PyTorch-style W[col][k], so B[k][n] = W[n][k]).
// lane (l, hi): N = col0 + l; elem e: K = kk + 16*hi + e -> one 32B load.
__device__ __forceinline__ bf16x16 load_b(const bf16_t* W, int ldk, int col0,
                                          int kk, int l, int hi) {
  return *(const bf16x16*)(W + (size_t)(col0 + l) * ldk + kk + 16 * hi);
}

__device__ __forceinline__ float sigm(float x) {
  return 1.0f / (1.0f + __expf(-x));
}
__device__ __forceinline__ float tanh_f(float x) {
  x = fminf(fmaxf(x, -15.0f), 15.0f);
  float e = __expf(2.0f * x);
  return (e - 1.0f) / (e + 1.0f);
}

// ---------------------------------------------------------------------------
// Prep kernels: pack fp32 weights -> bf16, concat, pad; combine biases.
// ---------------------------------------------------------------------------
__global__ void pack_w1(const float* __restrict__ wih,
                        const float* __restrict__ whh,
                        bf16_t* __restrict__ o) {
  int i = blockIdx.x * blockDim.x + threadIdx.x;
  if (i >= 2048 * K1) return;
  int n = i / K1, k = i - n * K1;
  float v = (k < HID + VSZ) ? wih[n * (HID + VSZ) + k]
                            : whh[n * HID + (k - HID - VSZ)];
  o[i] = (bf16_t)v;
}

__global__ void pack_w2(const float* __restrict__ wih,
                        const float* __restrict__ whh,
                        bf16_t* __restrict__ o) {
  int i = blockIdx.x * blockDim.x + threadIdx.x;
  if (i >= 512 * K2) return;
  int n = i / K2, k = i - n * K2;
  float v = (k < HID) ? wih[n * HID + k] : whh[n * KSZ + (k - HID)];
  o[i] = (bf16_t)v;
}

__global__ void pack_wout(const float* __restrict__ w, bf16_t* __restrict__ o) {
  int i = blockIdx.x * blockDim.x + threadIdx.x;
  if (i >= VPAD * K3) return;
  int n = i / K3, k = i - n * K3;
  float v = (n < VOCAB) ? w[n * K3 + k] : 0.0f;
  o[i] = (bf16_t)v;
}

__global__ void pack_emb(const float* __restrict__ e, bf16_t* __restrict__ o) {
  int i = blockIdx.x * blockDim.x + threadIdx.x;
  if (i >= VOCAB * HID) return;
  o[i] = (bf16_t)e[i];
}

__global__ void vec_add(const float* __restrict__ a, const float* __restrict__ b,
                        float* __restrict__ o, int n) {
  int i = blockIdx.x * blockDim.x + threadIdx.x;
  if (i < n) o[i] = a[i] + b[i];
}

// ---------------------------------------------------------------------------
// Persistent decoder kernel: 16 blocks x 512 threads, each block owns 16
// batch rows and runs all TLEN steps.
// ---------------------------------------------------------------------------
__global__ __launch_bounds__(512, 1)
void decoder_kernel(const float* __restrict__ key,
                    const float* __restrict__ values,
                    const int* __restrict__ lens,
                    const int* __restrict__ text,
                    const bf16_t* __restrict__ EMB,
                    const bf16_t* __restrict__ W1,
                    const bf16_t* __restrict__ W2,
                    const bf16_t* __restrict__ WO,
                    const float* __restrict__ B1,
                    const float* __restrict__ B2,
                    const float* __restrict__ BOUT,
                    float* __restrict__ out) {
  extern __shared__ char smem_raw[];
  float* h1  = (float*)smem_raw;          // 16x512
  float* c1  = h1 + 16 * HID;             // 16x512
  float* h2  = c1 + 16 * HID;             // 16x128
  float* c2  = h2 + 16 * KSZ;             // 16x128
  float* ctx = c2 + 16 * KSZ;             // 16x128
  float* g2  = ctx + 16 * KSZ;            // 16x512  (LSTM2 preactivations)
  float* g1  = g2 + 16 * HID;             // 16x1024 (LSTM1 i,f preactivations)
  float* eng = g1 + 16 * 1024;            // 16x400  (energy / probs)
  bf16_t* x1 = (bf16_t*)(eng + 16 * T_ENC);  // 16x1152
  bf16_t* x2 = x1 + 16 * K1;              // 16x640
  bf16_t* x3 = x2 + 16 * K2;              // 16x256

  const int tid  = threadIdx.x;
  const int lane = tid & 31;
  const int l    = lane & 15;
  const int hi   = lane >> 4;
  const int wave = tid >> 5;   // 0..15
  const int n0   = blockIdx.x * 16;

  // ---- init state ----
  for (int i = tid; i < 16 * HID; i += 512) { h1[i] = 0.0f; c1[i] = 0.0f; }
  for (int i = tid; i < 16 * KSZ; i += 512) { h2[i] = 0.0f; c2[i] = 0.0f; }
  __syncthreads();

  // ctx0 = values.mean(axis=0) ; one wave per row
  {
    const int m = wave;
    const int n = n0 + m;
    float ca0 = 0.f, ca1 = 0.f, ca2 = 0.f, ca3 = 0.f;
#pragma unroll 1
    for (int t = 0; t < T_ENC; ++t) {
      const float* vp = values + ((size_t)t * NB + n) * VSZ;
      ca0 += vp[lane]; ca1 += vp[lane + 32];
      ca2 += vp[lane + 64]; ca3 += vp[lane + 96];
    }
    const float s = 1.0f / (float)T_ENC;
    ctx[m * VSZ + lane]      = ca0 * s;
    ctx[m * VSZ + lane + 32] = ca1 * s;
    ctx[m * VSZ + lane + 64] = ca2 * s;
    ctx[m * VSZ + lane + 96] = ca3 * s;
  }
  __syncthreads();

#pragma unroll 1
  for (int step = 0; step < TLEN; ++step) {
    // ---- phase 1: build x1 = [emb | ctx | h1] and the h2 half of x2 ----
    for (int i = tid; i < 16 * K1; i += 512) {
      int m = i / K1, k = i - m * K1;
      bf16_t v;
      if (k < HID) {
        int tok = text[(n0 + m) * TLEN + step];
        v = EMB[(size_t)tok * HID + k];
      } else if (k < HID + VSZ) {
        v = (bf16_t)ctx[m * VSZ + (k - HID)];
      } else {
        v = (bf16_t)h1[m * HID + (k - HID - VSZ)];
      }
      x1[i] = v;
    }
    for (int i = tid; i < 16 * KSZ; i += 512) {
      int m = i >> 7, u = i & 127;
      x2[m * K2 + HID + u] = (bf16_t)h2[i];
    }
    __syncthreads();

    // ---- phase 2a: GEMM1 pass A (gates i, f) -> g1 LDS ----
    {
      f32x8 acc[4];
#pragma unroll
      for (int t = 0; t < 4; ++t) acc[t] = zero8();

#pragma unroll 1
      for (int kk = 0; kk < K1; kk += 32) {
        bf16x16 a = load_a(x1, K1, kk, l, hi);
#pragma unroll
        for (int g = 0; g < 2; ++g) {
#pragma unroll
          for (int j = 0; j < 2; ++j) {
            int col0 = g * HID + wave * 32 + j * 16;
            bf16x16 b = load_b(W1, K1, col0, kk, l, hi);
            acc[g * 2 + j] = wmma_bf(a, b, acc[g * 2 + j]);
          }
        }
      }
#pragma unroll
      for (int g = 0; g < 2; ++g) {
#pragma unroll
        for (int j = 0; j < 2; ++j) {
          int nh = wave * 32 + j * 16 + l;          // hidden unit 0..511
          float bb = B1[g * HID + nh];
#pragma unroll
          for (int r = 0; r < 8; ++r)
            g1[(r + 8 * hi) * 1024 + g * HID + nh] = acc[g * 2 + j][r] + bb;
        }
      }
    }

    // ---- phase 2b: GEMM1 pass B (gates g, o) + LSTM1 gate math ----
    {
      f32x8 acc[4];
#pragma unroll
      for (int t = 0; t < 4; ++t) acc[t] = zero8();

#pragma unroll 1
      for (int kk = 0; kk < K1; kk += 32) {
        bf16x16 a = load_a(x1, K1, kk, l, hi);
#pragma unroll
        for (int g = 0; g < 2; ++g) {
#pragma unroll
          for (int j = 0; j < 2; ++j) {
            int col0 = (2 + g) * HID + wave * 32 + j * 16;
            bf16x16 b = load_b(W1, K1, col0, kk, l, hi);
            acc[g * 2 + j] = wmma_bf(a, b, acc[g * 2 + j]);
          }
        }
      }
      // gate math: i,f from LDS (written by this same wave), g,o in regs
#pragma unroll
      for (int j = 0; j < 2; ++j) {
        int nh = wave * 32 + j * 16 + l;
        float bg = B1[2 * HID + nh];
        float bo = B1[3 * HID + nh];
#pragma unroll
        for (int r = 0; r < 8; ++r) {
          int m = r + 8 * hi;
          float iv = sigm(g1[m * 1024 + nh]);
          float fv = sigm(g1[m * 1024 + HID + nh]);
          float gv = tanh_f(acc[0 + j][r] + bg);
          float ov = sigm(acc[2 + j][r] + bo);
          float cn = fv * c1[m * HID + nh] + iv * gv;
          float hn = ov * tanh_f(cn);
          c1[m * HID + nh] = cn;
          h1[m * HID + nh] = hn;
          x2[m * K2 + nh]  = (bf16_t)hn;
        }
      }
    }
    __syncthreads();

    // ---- phase 3: GEMM2 (16x512 = x2 @ W2^T) -> g2, then LSTM2 gates ----
    {
      f32x8 a2[2];
#pragma unroll
      for (int t = 0; t < 2; ++t) a2[t] = zero8();

#pragma unroll 1
      for (int kk = 0; kk < K2; kk += 32) {
        bf16x16 a = load_a(x2, K2, kk, l, hi);
#pragma unroll
        for (int t = 0; t < 2; ++t) {
          int col0 = (wave * 2 + t) * 16;
          bf16x16 b = load_b(W2, K2, col0, kk, l, hi);
          a2[t] = wmma_bf(a, b, a2[t]);
        }
      }
#pragma unroll
      for (int t = 0; t < 2; ++t) {
        int col = (wave * 2 + t) * 16 + l;
        float bb = B2[col];
#pragma unroll
        for (int r = 0; r < 8; ++r)
          g2[(r + 8 * hi) * HID + col] = a2[t][r] + bb;
      }
    }
    __syncthreads();

    for (int i = tid; i < 16 * KSZ; i += 512) {
      int m = i >> 7, u = i & 127;
      float iv = sigm(g2[m * HID + u]);
      float fv = sigm(g2[m * HID + KSZ + u]);
      float gv = tanh_f(g2[m * HID + 2 * KSZ + u]);
      float ov = sigm(g2[m * HID + 3 * KSZ + u]);
      float cn = fv * c2[i] + iv * gv;
      float hn = ov * tanh_f(cn);
      c2[i] = cn;
      h2[i] = hn;
      x3[m * K3 + u] = (bf16_t)hn;
    }
    __syncthreads();

    // ---- phase 4: attention (one wave per batch row) ----
    {
      const int m = wave;
      const int n = n0 + m;
      const int len = lens[n];
      const float* h2r = h2 + m * KSZ;

      float mx = -3.0e38f;
#pragma unroll 1
      for (int t = lane; t < T_ENC; t += 32) {
        float e = -1.0e9f;
        if (t < len) {
          const float* kp = key + ((size_t)t * NB + n) * KSZ;
          float s = 0.f;
#pragma unroll
          for (int k = 0; k < KSZ; k += 4) {
            f32x4 kv = *(const f32x4*)(kp + k);
            s += kv[0] * h2r[k] + kv[1] * h2r[k + 1] +
                 kv[2] * h2r[k + 2] + kv[3] * h2r[k + 3];
          }
          e = s;
        }
        eng[m * T_ENC + t] = e;
        mx = fmaxf(mx, e);
      }
#pragma unroll
      for (int off = 16; off > 0; off >>= 1)
        mx = fmaxf(mx, __shfl_xor(mx, off, 32));

      float sum = 0.f;
#pragma unroll 1
      for (int t = lane; t < T_ENC; t += 32) {
        float p = __expf(eng[m * T_ENC + t] - mx);
        eng[m * T_ENC + t] = p;
        sum += p;
      }
#pragma unroll
      for (int off = 16; off > 0; off >>= 1)
        sum += __shfl_xor(sum, off, 32);
      float inv = 1.0f / sum;

      float ca0 = 0.f, ca1 = 0.f, ca2 = 0.f, ca3 = 0.f;
#pragma unroll 1
      for (int t = 0; t < len; ++t) {           // p==0 beyond len
        float p = eng[m * T_ENC + t];
        const float* vp = values + ((size_t)t * NB + n) * VSZ;
        ca0 += p * vp[lane];      ca1 += p * vp[lane + 32];
        ca2 += p * vp[lane + 64]; ca3 += p * vp[lane + 96];
      }
      float c0 = ca0 * inv, c1v = ca1 * inv, c2v = ca2 * inv, c3v = ca3 * inv;
      ctx[m * VSZ + lane]      = c0;
      ctx[m * VSZ + lane + 32] = c1v;
      ctx[m * VSZ + lane + 64] = c2v;
      ctx[m * VSZ + lane + 96] = c3v;
      x3[m * K3 + KSZ + lane]      = (bf16_t)c0;
      x3[m * K3 + KSZ + lane + 32] = (bf16_t)c1v;
      x3[m * K3 + KSZ + lane + 64] = (bf16_t)c2v;
      x3[m * K3 + KSZ + lane + 96] = (bf16_t)c3v;
    }
    __syncthreads();

    // ---- phase 5: GEMM3 (16x1008 = x3 @ WO^T) + bias -> logits ----
    {
      f32x8 a3[4];
#pragma unroll
      for (int q = 0; q < 4; ++q) a3[q] = zero8();

#pragma unroll 1
      for (int kk = 0; kk < K3; kk += 32) {
        bf16x16 a = load_a(x3, K3, kk, l, hi);
#pragma unroll
        for (int q = 0; q < 4; ++q) {
          int tt = wave + 16 * q;
          if (tt < VPAD / 16) {
            bf16x16 b = load_b(WO, K3, tt * 16, kk, l, hi);
            a3[q] = wmma_bf(a, b, a3[q]);
          }
        }
      }
#pragma unroll
      for (int q = 0; q < 4; ++q) {
        int tt = wave + 16 * q;
        if (tt >= VPAD / 16) continue;
        int col = tt * 16 + l;
        if (col < VOCAB) {
          float bb = BOUT[col];
#pragma unroll
          for (int r = 0; r < 8; ++r) {
            int m = r + 8 * hi;
            out[((size_t)(n0 + m) * TLEN + step) * VOCAB + col] = a3[q][r] + bb;
          }
        }
      }
    }
    __syncthreads();
  }
}

// ---------------------------------------------------------------------------
// Launcher
// ---------------------------------------------------------------------------
extern "C" void kernel_launch(void* const* d_in, const int* in_sizes, int n_in,
                              void* d_out, int out_size, void* d_ws,
                              size_t ws_size, hipStream_t stream) {
  const float* key    = (const float*)d_in[0];
  const float* values = (const float*)d_in[1];
  const int*   lens   = (const int*)d_in[2];
  const int*   text   = (const int*)d_in[3];
  const float* emb    = (const float*)d_in[4];
  const float* wih1   = (const float*)d_in[5];
  const float* whh1   = (const float*)d_in[6];
  const float* bih1   = (const float*)d_in[7];
  const float* bhh1   = (const float*)d_in[8];
  const float* wih2   = (const float*)d_in[9];
  const float* whh2   = (const float*)d_in[10];
  const float* bih2   = (const float*)d_in[11];
  const float* bhh2   = (const float*)d_in[12];
  const float* wout   = (const float*)d_in[13];
  const float* bout   = (const float*)d_in[14];

  char* ws = (char*)d_ws;
  bf16_t* W1 = (bf16_t*)ws; ws += (size_t)2048 * K1 * 2;
  bf16_t* W2 = (bf16_t*)ws; ws += (size_t)512 * K2 * 2;
  bf16_t* WO = (bf16_t*)ws; ws += (size_t)VPAD * K3 * 2;
  bf16_t* EM = (bf16_t*)ws; ws += (size_t)VOCAB * HID * 2;
  float*  B1 = (float*)ws;  ws += (size_t)2048 * 4;
  float*  B2 = (float*)ws;  ws += (size_t)512 * 4;

  pack_w1<<<(2048 * K1 + 255) / 256, 256, 0, stream>>>(wih1, whh1, W1);
  pack_w2<<<(512 * K2 + 255) / 256, 256, 0, stream>>>(wih2, whh2, W2);
  pack_wout<<<(VPAD * K3 + 255) / 256, 256, 0, stream>>>(wout, WO);
  pack_emb<<<(VOCAB * HID + 255) / 256, 256, 0, stream>>>(emb, EM);
  vec_add<<<(2048 + 255) / 256, 256, 0, stream>>>(bih1, bhh1, B1, 2048);
  vec_add<<<(512 + 255) / 256, 256, 0, stream>>>(bih2, bhh2, B2, 512);

  // LDS bytes: fp32 state (h1,c1,g1,g2,h2,c2,ctx,eng) + bf16 x1/x2/x3
  const size_t SMEM =
      (size_t)(16 * HID * 2 + 16 * KSZ * 3 + 16 * HID + 16 * 1024 +
               16 * T_ENC) * 4 +
      (size_t)16 * (K1 + K2 + K3) * 2;   // = 279552
  (void)hipFuncSetAttribute(reinterpret_cast<const void*>(decoder_kernel),
                            hipFuncAttributeMaxDynamicSharedMemorySize,
                            (int)SMEM);
  decoder_kernel<<<NB / 16, 512, SMEM, stream>>>(
      key, values, lens, text, EM, W1, W2, WO, B1, B2, bout, (float*)d_out);
}
